// MoeMMBlock_20298015441153
// MI455X (gfx1250) — matrix operand order repaired and checked by reading, not compiled, not run on hardware
//
#include <hip/hip_runtime.h>
#include <stdint.h>

// ---------------- constants (match reference shapes) ----------------
#define Bc   8
#define NQc  1024
#define NAc  512
#define NVc  512
#define NAVc 1024
#define Dc   2048
#define Hc   16
#define DHc  128
#define Rc   16
#define FFc  8192
#define CHc  1024   // MLP row-chunk

typedef unsigned short u16;
typedef __attribute__((ext_vector_type(16))) __bf16 v16bf;
typedef __attribute__((ext_vector_type(8)))  float  v8f;

union FragU { v16bf v; uint4 u[2]; };

__device__ __forceinline__ u16 f2bf(float f) {
  union { float f; uint32_t u; } c; c.f = f;
  uint32_t u = c.u;
  u += 0x7FFFu + ((u >> 16) & 1u);   // round-to-nearest-even
  return (u16)(u >> 16);
}

#define LDS_TILE (128 * 40)          // elements per buffer (padded stride 40)
#define LDS_TILE_BYTES (LDS_TILE * 2)

// ---------------- general bf16 WMMA GEMM ----------------
// C[M,N] = epi( alpha * A[M,K] @ B + bias[n] ), batched over grid.z.
// A: bf16 row-major, lda. B: bf16; bT=1 -> stored [N,K]; bT=0 -> stored [K,N].
// epi: v = outer * (alpha*acc + bias[n]); outer = (*aPtr) * gate[(m/rowsPerB)*4+eIdx]
// Cb!=null -> store bf16 (no accumulate); else Cf, accum? += : =
// Requires K % 8 == 0 (true for all uses here: 16,128,512,1024,2048,8192).
// Fast path (bT, full 128x128 tile, K%32==0): async global->LDS double buffering.
__global__ __launch_bounds__(256) void gemm_bf16_kernel(
    const u16* __restrict__ A, long lda, long aBS,
    const u16* __restrict__ Bm, long ldb, long bBS, int bT,
    float* __restrict__ Cf, u16* __restrict__ Cb, long ldc, long cBS,
    int M, int N, int K,
    const float* __restrict__ bias, float alpha, const float* __restrict__ aPtr,
    const float* __restrict__ gatep, int eIdx, int rowsPerB, int accum)
{
  __shared__ __align__(16) u16 As[2 * LDS_TILE];
  __shared__ __align__(16) u16 Bs[2 * LDS_TILE];
  const int tid  = threadIdx.x;
  const long bz  = blockIdx.z;
  A  += bz * aBS;
  Bm += bz * bBS;
  const long cOff = bz * cBS;
  const int mBase = blockIdx.y * 128;
  const int nBase = blockIdx.x * 128;
  const int wave = tid >> 5, lane = tid & 31;
  const int wm = wave & 3, wn = wave >> 2;      // 4 (M) x 2 (N) waves
  const int lr = lane & 15, lh = lane >> 4;

  v8f acc[2][4];
#pragma unroll
  for (int i = 0; i < 2; ++i)
#pragma unroll
    for (int j = 0; j < 4; ++j)
#pragma unroll
      for (int r = 0; r < 8; ++r) acc[i][j][r] = 0.0f;

  const int rowL = tid >> 2;          // 0..63
  const int kcL  = (tid & 3) << 3;    // 0,8,16,24

  const bool fast = bT && ((K & 31) == 0) && (mBase + 128 <= M) && (nBase + 128 <= N);

  if (fast) {
    // ---------- async double-buffered pipeline ----------
    const u16* pA0 = A  + (long)(mBase + rowL) * lda + kcL;
    const u16* pA1 = pA0 + 64 * lda;
    const u16* pB0 = Bm + (long)(nBase + rowL) * ldb + kcL;
    const u16* pB1 = pB0 + 64 * ldb;
    const unsigned dA0 = (unsigned)(unsigned long long)&As[rowL * 40 + kcL];
    const unsigned dA1 = dA0 + 64 * 40 * 2;
    const unsigned dB0 = (unsigned)(unsigned long long)&Bs[rowL * 40 + kcL];
    const unsigned dB1 = dB0 + 64 * 40 * 2;

    // prologue: tile 0 -> buffer 0
    asm volatile("global_load_async_to_lds_b128 %0, %1, off" :: "v"(dA0), "v"(pA0) : "memory");
    asm volatile("global_load_async_to_lds_b128 %0, %1, off" :: "v"(dA1), "v"(pA1) : "memory");
    asm volatile("global_load_async_to_lds_b128 %0, %1, off" :: "v"(dB0), "v"(pB0) : "memory");
    asm volatile("global_load_async_to_lds_b128 %0, %1, off" :: "v"(dB1), "v"(pB1) : "memory");
    pA0 += 32; pA1 += 32; pB0 += 32; pB1 += 32;

    unsigned cur = 0;                  // byte offset of current buffer (0 / LDS_TILE_BYTES)
    for (int kb = 0; kb < K; kb += 32) {
      asm volatile("s_wait_asynccnt 0x0" ::: "memory");  // this wave's tile data landed
      __syncthreads();                                   // all waves landed; prev buffer readers done
      if (kb + 32 < K) {
        unsigned nb = cur ^ LDS_TILE_BYTES;
        asm volatile("global_load_async_to_lds_b128 %0, %1, off" :: "v"(dA0 + nb), "v"(pA0) : "memory");
        asm volatile("global_load_async_to_lds_b128 %0, %1, off" :: "v"(dA1 + nb), "v"(pA1) : "memory");
        asm volatile("global_load_async_to_lds_b128 %0, %1, off" :: "v"(dB0 + nb), "v"(pB0) : "memory");
        asm volatile("global_load_async_to_lds_b128 %0, %1, off" :: "v"(dB1 + nb), "v"(pB1) : "memory");
        pA0 += 32; pA1 += 32; pB0 += 32; pB1 += 32;
      }
      const u16* asB = As + (cur >> 1);
      const u16* bsB = Bs + (cur >> 1);
      FragU af[2], bfr[4];
#pragma unroll
      for (int i = 0; i < 2; ++i) {
        // A 16-bit layout: lanes0-15 K=0..7,16..23 ; lanes16-31 K=8..15,24..31
        const u16* p = &asB[(wm * 32 + i * 16 + lr) * 40 + lh * 8];
        af[i].u[0] = *(const uint4*)p;
        af[i].u[1] = *(const uint4*)(p + 16);
      }
#pragma unroll
      for (int j = 0; j < 4; ++j) {
        // B 16-bit layout: lanes0-15 K=0..7,16..23 ; lanes16-31 K=8..15,24..31
        const u16* p = &bsB[(wn * 64 + j * 16 + lr) * 40 + lh * 8];
        bfr[j].u[0] = *(const uint4*)p;
        bfr[j].u[1] = *(const uint4*)(p + 16);
      }
#pragma unroll
      for (int i = 0; i < 2; ++i)
#pragma unroll
        for (int j = 0; j < 4; ++j)
          acc[i][j] = __builtin_amdgcn_wmma_f32_16x16x32_bf16(
              false, af[i].v, false, bfr[j].v, (short)0, acc[i][j], false, false);
      cur ^= LDS_TILE_BYTES;
    }
  } else {
    // ---------- guarded single-buffer path (partial tiles / B stored [K,N]) ----------
    for (int kb = 0; kb < K; kb += 32) {
#pragma unroll
      for (int it = 0; it < 2; ++it) {
        int i = tid + it * 256;
        int row = i >> 2, kc = (i & 3) << 3;
        int gm = mBase + row, gk = kb + kc;
        uint4 val = make_uint4(0u, 0u, 0u, 0u);
        if (gm < M && gk + 8 <= K) val = *(const uint4*)(A + (long)gm * lda + gk);
        *(uint4*)&As[row * 40 + kc] = val;
      }
      if (bT) {
#pragma unroll
        for (int it = 0; it < 2; ++it) {
          int i = tid + it * 256;
          int row = i >> 2, kc = (i & 3) << 3;
          int gn = nBase + row, gk = kb + kc;
          uint4 val = make_uint4(0u, 0u, 0u, 0u);
          if (gn < N && gk + 8 <= K) val = *(const uint4*)(Bm + (long)gn * ldb + gk);
          *(uint4*)&Bs[row * 40 + kc] = val;
        }
      } else {
        for (int i = tid; i < 4096; i += 256) {
          int k = i >> 7, n = i & 127;
          int gk = kb + k, gn = nBase + n;
          u16 v = 0;
          if (gk < K && gn < N) v = Bm[(long)gk * ldb + gn];
          Bs[n * 40 + k] = v;
        }
      }
      __syncthreads();
      FragU af[2], bfr[4];
#pragma unroll
      for (int i = 0; i < 2; ++i) {
        const u16* p = &As[(wm * 32 + i * 16 + lr) * 40 + lh * 8];
        af[i].u[0] = *(const uint4*)p;
        af[i].u[1] = *(const uint4*)(p + 16);
      }
#pragma unroll
      for (int j = 0; j < 4; ++j) {
        const u16* p = &Bs[(wn * 64 + j * 16 + lr) * 40 + lh * 8];
        bfr[j].u[0] = *(const uint4*)p;
        bfr[j].u[1] = *(const uint4*)(p + 16);
      }
#pragma unroll
      for (int i = 0; i < 2; ++i)
#pragma unroll
        for (int j = 0; j < 4; ++j)
          acc[i][j] = __builtin_amdgcn_wmma_f32_16x16x32_bf16(
              false, af[i].v, false, bfr[j].v, (short)0, acc[i][j], false, false);
      __syncthreads();
    }
  }

  // ---- epilogue ----
  float oBase = (aPtr ? aPtr[0] : 1.0f);
#pragma unroll
  for (int i = 0; i < 2; ++i) {
    int mSub = mBase + wm * 32 + i * 16 + lh * 8;
#pragma unroll
    for (int j = 0; j < 4; ++j) {
      int n = nBase + wn * 64 + j * 16 + lr;
#pragma unroll
      for (int r = 0; r < 8; ++r) {
        int m = mSub + r;
        if (m < M && n < N) {
          float v = alpha * acc[i][j][r] + (bias ? bias[n] : 0.0f);
          float s = oBase * (gatep ? gatep[(m / rowsPerB) * 4 + eIdx] : 1.0f);
          v *= s;
          long ci = cOff + (long)m * ldc + n;
          if (Cb) Cb[ci] = f2bf(v);
          else    Cf[ci] = accum ? (Cf[ci] + v) : v;
        }
      }
    }
  }
}

// ---------------- conversions ----------------
__global__ void cvt_kernel(const float* __restrict__ src, u16* __restrict__ dst, long n) {
  long stride = (long)gridDim.x * blockDim.x;
  for (long i = (long)blockIdx.x * blockDim.x + threadIdx.x; i < n; i += stride)
    dst[i] = f2bf(src[i]);
}

// src[K,N] (f32) -> dst[N,K] (bf16); batched via blockIdx.y
__global__ void cvtT_kernel(const float* __restrict__ src, u16* __restrict__ dst, int K, int N) {
  long total = (long)K * N;
  long base = (long)blockIdx.y * total;
  long stride = (long)gridDim.x * blockDim.x;
  for (long i = (long)blockIdx.x * blockDim.x + threadIdx.x; i < total; i += stride) {
    int k = (int)(i % K);
    long n = i / K;
    dst[base + i] = f2bf(src[base + (long)k * N + n]);
  }
}

// ---------------- router ----------------
__global__ void rms_rows_kernel(const float* __restrict__ x, float* __restrict__ rrms, int D) {
  long row = blockIdx.x;
  const float* xr = x + row * (long)D;
  __shared__ float red[256];
  float s = 0.f;
  for (int i = threadIdx.x; i < D; i += 256) { float v = xr[i]; s += v * v; }
  red[threadIdx.x] = s; __syncthreads();
  for (int o = 128; o > 0; o >>= 1) { if (threadIdx.x < o) red[threadIdx.x] += red[threadIdx.x + o]; __syncthreads(); }
  if (threadIdx.x == 0) rrms[row] = rsqrtf(red[0] / D + 1e-6f);
}

__global__ void router_mean_kernel(const float* __restrict__ x, const float* __restrict__ rrms,
                                   const float* __restrict__ w, float* __restrict__ m, int NQ, int D) {
  int d = blockIdx.x * blockDim.x + threadIdx.x;
  int b = blockIdx.y;
  if (d >= D) return;
  const float* xb = x + (long)b * NQ * D + d;
  const float* rb = rrms + (long)b * NQ;
  float s = 0.f;
  for (int n = 0; n < NQ; ++n) s += xb[(long)n * D] * rb[n];
  m[(long)b * D + d] = s * w[d] / (float)NQ;
}

__global__ void router_finalize_kernel(const float* __restrict__ m, const float* __restrict__ W,
                                       const float* __restrict__ rb, const float* __restrict__ a1,
                                       const float* __restrict__ a2, float* __restrict__ gate,
                                       float* __restrict__ scal, int D, int B) {
  __shared__ float sl[32];
  int t = threadIdx.x;
  if (t < B * 4) {
    int b = t >> 2, e = t & 3;
    const float* mb = m + (long)b * D;
    float s = 0.f;
    for (int d = 0; d < D; ++d) s += mb[d] * W[d * 4 + e];
    sl[t] = s + rb[e];
  }
  __syncthreads();
  if (t < B) {
    float l[4], w[4];
    float mx = -1e30f;
    for (int e = 0; e < 4; ++e) { l[e] = sl[t * 4 + e]; mx = fmaxf(mx, l[e]); }
    float sum = 0.f;
    for (int e = 0; e < 4; ++e) { w[e] = expf(l[e] - mx); sum += w[e]; }
    for (int e = 0; e < 4; ++e) w[e] /= sum;
    int i1 = 0;
    for (int e = 1; e < 4; ++e) if (w[e] > w[i1]) i1 = e;
    int i2 = -1;
    for (int e = 0; e < 4; ++e) if (e != i1 && (i2 < 0 || w[e] > w[i2])) i2 = e;
    float inv = 1.f / (w[i1] + w[i2] + 1e-10f);
    float g[4] = {0.f, 0.f, 0.f, 0.f};
    g[i1] = w[i1] * inv; g[i2] = w[i2] * inv;
    for (int e = 0; e < 4; ++e) gate[t * 4 + e] = g[e];
  }
  if (t == 0) {
    scal[0] = 1.f / (1.f + expf(-a1[0]));
    scal[1] = 1.f / (1.f + expf(-a2[0]));
  }
}

// ---------------- attention softmax (rows of S -> bf16 P) ----------------
__global__ void softmax_rows_kernel(const float* __restrict__ S, u16* __restrict__ P, int NK) {
  long row = blockIdx.x;
  const float* sr = S + row * (long)NK;
  u16* pr = P + row * (long)NK;
  __shared__ float red[256];
  int t = threadIdx.x;
  float mx = -1e30f;
  for (int i = t; i < NK; i += 256) mx = fmaxf(mx, sr[i]);
  red[t] = mx; __syncthreads();
  for (int o = 128; o > 0; o >>= 1) { if (t < o) red[t] = fmaxf(red[t], red[t + o]); __syncthreads(); }
  float rowMax = red[0]; __syncthreads();
  float s = 0.f;
  for (int i = t; i < NK; i += 256) s += expf(sr[i] - rowMax);
  red[t] = s; __syncthreads();
  for (int o = 128; o > 0; o >>= 1) { if (t < o) red[t] += red[t + o]; __syncthreads(); }
  float inv = 1.f / red[0];
  for (int i = t; i < NK; i += 256) pr[i] = f2bf(expf(sr[i] - rowMax) * inv);
}

// ---------------- elementwise ----------------
__global__ void x1_combine_kernel(const float* __restrict__ xq, float* __restrict__ delta_x1,
                                  float* __restrict__ outF, const float* __restrict__ scal, long n) {
  long stride = (long)gridDim.x * blockDim.x;
  float s1 = scal[0];
  for (long i = (long)blockIdx.x * blockDim.x + threadIdx.x; i < n; i += stride) {
    float v = xq[i] + s1 * delta_x1[i];
    delta_x1[i] = v;   // becomes x1
    outF[i] = v;       // seed residual in d_out
  }
}

__global__ void rmsnorm_rows_bf16_kernel(const float* __restrict__ x, const float* __restrict__ w,
                                         u16* __restrict__ out, int D) {
  long row = blockIdx.x;
  const float* xr = x + row * (long)D;
  u16* orow = out + row * (long)D;
  __shared__ float red[256];
  __shared__ float rrS;
  float s = 0.f;
  for (int i = threadIdx.x; i < D; i += 256) { float v = xr[i]; s += v * v; }
  red[threadIdx.x] = s; __syncthreads();
  for (int o = 128; o > 0; o >>= 1) { if (threadIdx.x < o) red[threadIdx.x] += red[threadIdx.x + o]; __syncthreads(); }
  if (threadIdx.x == 0) rrS = rsqrtf(red[0] / D + 1e-6f);
  __syncthreads();
  float rr = rrS;
  for (int i = threadIdx.x; i < D; i += 256) orow[i] = f2bf(w[i] * xr[i] * rr);
}

__global__ void silu_comb_kernel(const float* __restrict__ xg, const float* __restrict__ xu,
                                 u16* __restrict__ xd, long n) {
  long stride = (long)gridDim.x * blockDim.x;
  for (long i = (long)blockIdx.x * blockDim.x + threadIdx.x; i < n; i += stride) {
    float g = xg[i];
    float sg = g / (1.f + expf(-g));   // silu
    xd[i] = f2bf(sg + xu[i]);
  }
}

// ---------------- host ----------------
extern "C" void kernel_launch(void* const* d_in, const int* in_sizes, int n_in,
                              void* d_out, int out_size, void* d_ws, size_t ws_size,
                              hipStream_t stream) {
  (void)in_sizes; (void)n_in; (void)out_size; (void)ws_size;
  const float* x_q    = (const float*)d_in[0];
  const float* z_a    = (const float*)d_in[1];
  const float* z_v    = (const float*)d_in[2];
  const float* z_av   = (const float*)d_in[3];
  const float* ln1_w  = (const float*)d_in[4];
  const float* router_W = (const float*)d_in[5];
  const float* router_b = (const float*)d_in[6];
  const float* Wq = (const float*)d_in[7];
  const float* bq = (const float*)d_in[8];
  const float* Wk = (const float*)d_in[9];
  const float* bk = (const float*)d_in[10];
  const float* Wv = (const float*)d_in[11];
  const float* bv = (const float*)d_in[12];
  const float* Wo = (const float*)d_in[13];
  const float* bo = (const float*)d_in[14];
  const float* alpha1 = (const float*)d_in[15];
  const float* alpha2 = (const float*)d_in[16];
  const float* ln2_w  = (const float*)d_in[17];
  const float* gate_w = (const float*)d_in[18];
  const float* up_w   = (const float*)d_in[19];
  const float* down_w = (const float*)d_in[20];
  const float* la_gA  = (const float*)d_in[21];
  const float* la_gB  = (const float*)d_in[22];
  const float* la_uA  = (const float*)d_in[23];
  const float* la_uB  = (const float*)d_in[24];
  const float* la_dA  = (const float*)d_in[25];
  const float* la_dB  = (const float*)d_in[26];
  float* outF = (float*)d_out;

  char* ws = (char*)d_ws;
  size_t off = 0;
  auto alloc = [&](size_t bytes) -> void* {
    void* p = ws + off;
    off += (bytes + 255) & ~(size_t)255;
    return p;
  };
  // weights (bf16, transposed to [N,K])
  u16* WQT   = (u16*)alloc((size_t)3 * Dc * Dc * 2);
  u16* WKT   = (u16*)alloc((size_t)3 * Dc * Dc * 2);
  u16* WVT   = (u16*)alloc((size_t)3 * Dc * Dc * 2);
  u16* WOT   = (u16*)alloc((size_t)3 * Dc * Dc * 2);
  u16* GATET = (u16*)alloc((size_t)FFc * Dc * 2);
  u16* UPT   = (u16*)alloc((size_t)FFc * Dc * 2);
  u16* DOWNT = (u16*)alloc((size_t)Dc * FFc * 2);
  u16* LGA   = (u16*)alloc((size_t)Rc * Dc * 2);
  u16* LGB   = (u16*)alloc((size_t)FFc * Rc * 2);
  u16* LUA   = (u16*)alloc((size_t)Rc * Dc * 2);
  u16* LUB   = (u16*)alloc((size_t)FFc * Rc * 2);
  u16* LDAT  = (u16*)alloc((size_t)Rc * FFc * 2);
  u16* LDBT  = (u16*)alloc((size_t)Dc * Rc * 2);
  // activations
  u16* XQB = (u16*)alloc((size_t)Bc * NQc * Dc * 2);
  u16* ZB  = (u16*)alloc((size_t)Bc * NAVc * Dc * 2);
  u16* QB  = (u16*)alloc((size_t)Bc * NQc * Dc * 2);
  u16* KB  = (u16*)alloc((size_t)Bc * NAVc * Dc * 2);
  u16* VB  = (u16*)alloc((size_t)Bc * NAVc * Dc * 2);
  u16* OB  = (u16*)alloc((size_t)Bc * NQc * Dc * 2);
  float* S = (float*)alloc((size_t)Hc * NQc * NAVc * 4);
  u16* P   = (u16*)alloc((size_t)Hc * NQc * NAVc * 2);
  float* DELTA = (float*)alloc((size_t)Bc * NQc * Dc * 4);  // later holds x1
  u16* HB  = (u16*)alloc((size_t)Bc * NQc * Dc * 2);
  float* XG = (float*)alloc((size_t)CHc * FFc * 4);
  float* XU = (float*)alloc((size_t)CHc * FFc * 4);
  u16* XDB = (u16*)alloc((size_t)CHc * FFc * 2);
  u16* T1  = (u16*)alloc((size_t)CHc * Rc * 2);
  u16* T1U = (u16*)alloc((size_t)CHc * Rc * 2);
  u16* T2  = (u16*)alloc((size_t)CHc * Rc * 2);
  float* MROUT = (float*)alloc((size_t)Bc * Dc * 4);
  float* RRMS  = (float*)alloc((size_t)Bc * NQc * 4);
  float* GATEW = (float*)alloc((size_t)Bc * 4 * 4);
  float* SCAL  = (float*)alloc(2 * 4);

  auto gemm = [&](const u16* A, long lda, long aBS,
                  const u16* Bm, long ldb, long bBS, int bT,
                  float* Cf, u16* Cb, long ldc, long cBS,
                  int M, int N, int K,
                  const float* bias, float alpha, const float* aPtr,
                  const float* gatep, int eIdx, int rowsPerB, int accum, int batch) {
    dim3 g((N + 127) / 128, (M + 127) / 128, batch);
    gemm_bf16_kernel<<<g, dim3(256), 0, stream>>>(A, lda, aBS, Bm, ldb, bBS, bT,
        Cf, Cb, ldc, cBS, M, N, K, bias, alpha, aPtr, gatep, eIdx, rowsPerB, accum);
  };
  auto cvt = [&](const float* src, u16* dst, long n) {
    cvt_kernel<<<dim3(4096), dim3(256), 0, stream>>>(src, dst, n);
  };
  auto cvtT = [&](const float* src, u16* dst, int K, int N, int batch) {
    cvtT_kernel<<<dim3(2048, batch), dim3(256), 0, stream>>>(src, dst, K, N);
  };

  // ---- router ----
  rms_rows_kernel<<<dim3(Bc * NQc), dim3(256), 0, stream>>>(x_q, RRMS, Dc);
  router_mean_kernel<<<dim3(Dc / 256, Bc), dim3(256), 0, stream>>>(x_q, RRMS, ln1_w, MROUT, NQc, Dc);
  router_finalize_kernel<<<dim3(1), dim3(64), 0, stream>>>(MROUT, router_W, router_b,
                                                           alpha1, alpha2, GATEW, SCAL, Dc, Bc);
  // ---- weight / activation conversion ----
  cvt(x_q, XQB, (long)Bc * NQc * Dc);
  cvtT(Wq, WQT, Dc, Dc, 3);
  cvtT(Wk, WKT, Dc, Dc, 3);
  cvtT(Wv, WVT, Dc, Dc, 3);
  cvtT(Wo, WOT, Dc, Dc, 3);
  cvtT(gate_w, GATET, Dc, FFc, 1);
  cvtT(up_w,   UPT,   Dc, FFc, 1);
  cvtT(down_w, DOWNT, FFc, Dc, 1);
  cvtT(la_gA, LGA, Dc, Rc, 1);
  cvtT(la_gB, LGB, Rc, FFc, 1);
  cvtT(la_uA, LUA, Dc, Rc, 1);
  cvtT(la_uB, LUB, Rc, FFc, 1);
  cvtT(la_dA, LDAT, FFc, Rc, 1);
  cvtT(la_dB, LDBT, Rc, Dc, 1);

  // ---- experts ----
  const int   NKe[3]   = {NAc, NVc, NAVc};
  const float* zsrc[3] = {z_a, z_v, z_av};
  const float qkScale = 0.08838834764831845f;  // 1/sqrt(128)
  for (int e = 0; e < 3; ++e) {
    int NK = NKe[e];
    cvt(zsrc[e], ZB, (long)Bc * NK * Dc);
    gemm(XQB, Dc, 0, WQT + (long)e * Dc * Dc, Dc, 0, 1, nullptr, QB, Dc, 0,
         Bc * NQc, Dc, Dc, bq + e * Dc, 1.f, nullptr, nullptr, 0, 1, 0, 1);
    gemm(ZB, Dc, 0, WKT + (long)e * Dc * Dc, Dc, 0, 1, nullptr, KB, Dc, 0,
         Bc * NK, Dc, Dc, bk + e * Dc, 1.f, nullptr, nullptr, 0, 1, 0, 1);
    gemm(ZB, Dc, 0, WVT + (long)e * Dc * Dc, Dc, 0, 1, nullptr, VB, Dc, 0,
         Bc * NK, Dc, Dc, bv + e * Dc, 1.f, nullptr, nullptr, 0, 1, 0, 1);
    for (int b = 0; b < Bc; ++b) {
      // S[h,q,k] = Q_h K_h^T / sqrt(dh)   (batched over H heads)
      gemm(QB + (long)b * NQc * Dc, Dc, DHc,
           KB + (long)b * NK * Dc, Dc, DHc, 1,
           S, nullptr, NK, (long)NQc * NK,
           NQc, NK, DHc, nullptr, qkScale, nullptr, nullptr, 0, 1, 0, Hc);
      softmax_rows_kernel<<<dim3(Hc * NQc), dim3(256), 0, stream>>>(S, P, NK);
      // O_h = P_h V_h  (V sliced [NK, dh] from [NK, D], non-transposed B)
      gemm(P, NK, (long)NQc * NK,
           VB + (long)b * NK * Dc, Dc, DHc, 0,
           nullptr, OB + (long)b * NQc * Dc, Dc, DHc,
           NQc, DHc, NK, nullptr, 1.f, nullptr, nullptr, 0, 1, 0, Hc);
    }
    // delta (+)= gate[b,e] * (O @ Wo[e] + bo[e])
    gemm(OB, Dc, 0, WOT + (long)e * Dc * Dc, Dc, 0, 1, DELTA, nullptr, Dc, 0,
         Bc * NQc, Dc, Dc, bo + e * Dc, 1.f, nullptr, GATEW, e, NQc, (e > 0) ? 1 : 0, 1);
  }

  // ---- residual + ln2 ----
  long totalX = (long)Bc * NQc * Dc;
  x1_combine_kernel<<<dim3(8192), dim3(256), 0, stream>>>(x_q, DELTA, outF, SCAL, totalX);
  rmsnorm_rows_bf16_kernel<<<dim3(Bc * NQc), dim3(256), 0, stream>>>(DELTA, ln2_w, HB, Dc);

  // ---- MLP in row chunks ----
  for (int c = 0; c < (Bc * NQc) / CHc; ++c) {
    const u16* Ac = HB + (long)c * CHc * Dc;
    float* Oc = outF + (long)c * CHc * Dc;
    gemm(Ac, Dc, 0, LGA, Dc, 0, 1, nullptr, T1, Rc, 0, CHc, Rc, Dc,
         nullptr, 1.f, nullptr, nullptr, 0, 1, 0, 1);
    gemm(Ac, Dc, 0, GATET, Dc, 0, 1, XG, nullptr, FFc, 0, CHc, FFc, Dc,
         nullptr, 1.f, nullptr, nullptr, 0, 1, 0, 1);
    gemm(T1, Rc, 0, LGB, Rc, 0, 1, XG, nullptr, FFc, 0, CHc, FFc, Rc,
         nullptr, 1.f, nullptr, nullptr, 0, 1, 1, 1);
    gemm(Ac, Dc, 0, LUA, Dc, 0, 1, nullptr, T1U, Rc, 0, CHc, Rc, Dc,
         nullptr, 1.f, nullptr, nullptr, 0, 1, 0, 1);
    gemm(Ac, Dc, 0, UPT, Dc, 0, 1, XU, nullptr, FFc, 0, CHc, FFc, Dc,
         nullptr, 1.f, nullptr, nullptr, 0, 1, 0, 1);
    gemm(T1U, Rc, 0, LUB, Rc, 0, 1, XU, nullptr, FFc, 0, CHc, FFc, Rc,
         nullptr, 1.f, nullptr, nullptr, 0, 1, 1, 1);
    silu_comb_kernel<<<dim3(4096), dim3(256), 0, stream>>>(XG, XU, XDB, (long)CHc * FFc);
    // out += sig2 * (xd @ down_w)
    gemm(XDB, FFc, 0, DOWNT, FFc, 0, 1, Oc, nullptr, Dc, 0, CHc, Dc, FFc,
         nullptr, 1.f, SCAL + 1, nullptr, 0, 1, 1, 1);
    gemm(XDB, FFc, 0, LDAT, FFc, 0, 1, nullptr, T2, Rc, 0, CHc, Rc, FFc,
         nullptr, 1.f, nullptr, nullptr, 0, 1, 0, 1);
    // out += sig2 * (t2 @ la_dB)
    gemm(T2, Rc, 0, LDBT, Rc, 0, 1, Oc, nullptr, Dc, 0, CHc, Dc, Rc,
         nullptr, 1.f, SCAL + 1, nullptr, 0, 1, 1, 1);
  }
}
